// MultiHeadAttention_16647293239928
// MI455X (gfx1250) — matrix operand order repaired and checked by reading, not compiled, not run on hardware
//
#include <hip/hip_runtime.h>
#include <hip/hip_bf16.h>

// ---------------------------------------------------------------------------
// Whisper-style MHA for MI455X (gfx1250, wave32, WMMA f32_16x16x32_f16).
// Outputs: d_out = [ out (B*S*D fp32) | qk (B*H*S*S fp32, pre-softmax) ].
// - All LDS fragment reads are contiguous 16B/32B chunks -> ds_load_b128.
// - Guard-free, branch-free global->LDS staging (fixed-trip loops, buffers
//   over-allocated by one 64-row tile; junk rows discarded by epilogue guard).
// - K-step 64: 8 WMMAs per barrier pair.
// ---------------------------------------------------------------------------

typedef _Float16 f16;
typedef __attribute__((ext_vector_type(8)))  _Float16 v8h;
typedef __attribute__((ext_vector_type(16))) _Float16 v16h;
typedef __attribute__((ext_vector_type(8)))  float    v8f;

#define WMMA_F16(a, b, c) \
  __builtin_amdgcn_wmma_f32_16x16x32_f16(false, (a), false, (b), (short)0, (c), false, false)

__device__ __forceinline__ v16h cat16(v8h lo, v8h hi) {
  return __builtin_shufflevector(lo, hi, 0, 1, 2, 3, 4, 5, 6, 7,
                                 8, 9, 10, 11, 12, 13, 14, 15);
}

namespace cfg {
constexpr int B  = 4;
constexpr int S  = 1500;
constexpr int D  = 1280;
constexpr int H  = 20;
constexpr int HD = 64;
constexpr int BS = B * S;          // 6000 rows
constexpr int SP = 1536;           // padded key stride (mult of 64)
constexpr int PAD_ROWS = 64;       // over-allocation so staging needs no guards
constexpr float NEG   = -1e9f;
constexpr float SCALE = 0.35355339059327379f;  // 64^-0.25
}

// ---------------------------------------------------------------------------
// fp32 -> f16 elementwise convert
// ---------------------------------------------------------------------------
__global__ void k_f32_to_f16(const float* __restrict__ src, f16* __restrict__ dst, int n) {
  int i = blockIdx.x * blockDim.x + threadIdx.x;
  if (i < n) dst[i] = (f16)src[i];
}

// ---------------------------------------------------------------------------
// vT[b,h,d,s] = v[b,s,h,d], s padded to SP with zeros (tail-free wv K-loop).
// ---------------------------------------------------------------------------
__global__ void k_transpose_v(const f16* __restrict__ vh, f16* __restrict__ vT) {
  using namespace cfg;
  size_t i = (size_t)blockIdx.x * blockDim.x + threadIdx.x;
  size_t total = (size_t)B * H * HD * SP;
  if (i >= total) return;
  int s = (int)(i % SP);
  size_t r = i / SP;
  int d = (int)(r % HD);
  size_t bh = r / HD;
  int b = (int)(bh / H), h = (int)(bh % H);
  f16 val = (f16)0.f;
  if (s < S) val = vh[(((size_t)b * S + s) * H + h) * HD + d];
  vT[i] = val;
}

// ---------------------------------------------------------------------------
// C = (A @ W^T + bias) * scale.  A:[M(+pad),K] f16, W:[N,K] f16 (row-major).
// 4 waves / block (blockDim MUST be 128), tile 64x64, K step 64.
// ---------------------------------------------------------------------------
__global__ __launch_bounds__(128)
void k_gemm_xWt(const f16* __restrict__ A, const f16* __restrict__ W,
                const float* __restrict__ bias, float scale,
                f16* __restrict__ out_h, float* __restrict__ out_f,
                int M, int N, int K) {
  __shared__ f16 As[64][80];   // [m][k]  (160B stride: 16B/32B aligned chunks)
  __shared__ f16 Ws[64][80];   // [n][k]

  const int tid  = threadIdx.x;
  const int lane = tid & 31;
  const int wid  = tid >> 5;
  const int m0 = blockIdx.x * 64;
  const int n0 = blockIdx.y * 64;

  v8f acc[4];
#pragma unroll
  for (int t = 0; t < 4; ++t) acc[t] = (v8f){0.f, 0.f, 0.f, 0.f, 0.f, 0.f, 0.f, 0.f};

  const int kb  = (lane & 16) ? 8 : 0;    // unused: A frags read at kbb too (contiguous k)
  const int kbb = (lane & 16) ? 16 : 0;   // B-frag chunk base
  const int arow = wid * 16 + (lane & 15);
  const int ncl  = lane & 15;

  for (int k0 = 0; k0 < K; k0 += 64) {
    // stage A and W tiles: each 64 rows x 64 halves; 4 fixed b128 trips
#pragma unroll
    for (int it = 0; it < 4; ++it) {
      int idx = tid + it * 128;
      int r = idx >> 3, c = idx & 7;           // c: 8-half (16B) chunk
      *(uint4*)&As[r][8 * c] = *(const uint4*)(A + (size_t)(m0 + r) * K + k0 + 8 * c);
      *(uint4*)&Ws[r][8 * c] = *(const uint4*)(W + (size_t)(n0 + r) * K + k0 + 8 * c);
    }
    __syncthreads();

#pragma unroll
    for (int step = 0; step < 2; ++step) {
      int ko = step * 32;
      v16h a = cat16(*(const v8h*)&As[arow][ko + kbb],
                     *(const v8h*)&As[arow][ko + kbb + 8]);
#pragma unroll
      for (int t = 0; t < 4; ++t) {
        v16h bf = *(const v16h*)&Ws[t * 16 + ncl][ko + kbb];
        acc[t] = WMMA_F16(a, bf, acc[t]);
      }
    }
    __syncthreads();
  }

  const int mbase = m0 + wid * 16 + ((lane & 16) ? 8 : 0);
#pragma unroll
  for (int t = 0; t < 4; ++t) {
    int gn = n0 + t * 16 + ncl;
    float bv = bias ? bias[gn] : 0.0f;
#pragma unroll
    for (int r = 0; r < 8; ++r) {
      int gm = mbase + r;
      if (gm < M) {
        float v = (acc[t][r] + bv) * scale;
        if (out_f) out_f[(size_t)gm * N + gn] = v;
        else       out_h[(size_t)gm * N + gn] = (f16)v;
      }
    }
  }
}

// ---------------------------------------------------------------------------
// qk[b,h,i,j] = sum_d q[b,i,h,d]*k[b,j,h,d] + (j>i ? NEG : 0)  (fp32 out)
// grid (ktile, qtile, b*H); 4 waves/block; 64x64 score tile; hd=64 in LDS.
// ---------------------------------------------------------------------------
__global__ __launch_bounds__(128)
void k_attn_qk(const f16* __restrict__ qh, const f16* __restrict__ kh,
               float* __restrict__ qk_out) {
  using namespace cfg;
  __shared__ f16 Qs[64][80];   // [qrow][d]  (160B stride)
  __shared__ f16 Ks[64][80];   // [krow][d]

  const int tid  = threadIdx.x;
  const int lane = tid & 31;
  const int wid  = tid >> 5;
  const int k0 = blockIdx.x * 64;
  const int q0 = blockIdx.y * 64;
  const int bh = blockIdx.z;
  const int b  = bh / H, h = bh % H;

  const f16* qrow0 = qh + (((size_t)b * S + q0) * H + h) * HD;
  const f16* krow0 = kh + (((size_t)b * S + k0) * H + h) * HD;

  // stage Q and K tiles: 64 rows x 64 halves; 4 fixed b128 trips each
#pragma unroll
  for (int it = 0; it < 4; ++it) {
    int idx = tid + it * 128;
    int r = idx >> 3, c = idx & 7;
    *(uint4*)&Qs[r][8 * c] = *(const uint4*)(qrow0 + (size_t)r * (H * HD) + 8 * c);
    *(uint4*)&Ks[r][8 * c] = *(const uint4*)(krow0 + (size_t)r * (H * HD) + 8 * c);
  }
  __syncthreads();

  v8f acc[4];
#pragma unroll
  for (int t = 0; t < 4; ++t) acc[t] = (v8f){0.f, 0.f, 0.f, 0.f, 0.f, 0.f, 0.f, 0.f};

  const int kb  = (lane & 16) ? 8 : 0;
  const int kbb = (lane & 16) ? 16 : 0;
  const int arow = wid * 16 + (lane & 15);
  const int ncl  = lane & 15;

#pragma unroll
  for (int step = 0; step < 2; ++step) {
    int ko = step * 32;
    v16h a = cat16(*(const v8h*)&Qs[arow][ko + kbb],
                   *(const v8h*)&Qs[arow][ko + kbb + 8]);
#pragma unroll
    for (int t = 0; t < 4; ++t) {
      v16h bf = *(const v16h*)&Ks[t * 16 + ncl][ko + kbb];
      acc[t] = WMMA_F16(a, bf, acc[t]);
    }
  }

  float* base = qk_out + (size_t)bh * S * S;
  const int ibase = q0 + wid * 16 + ((lane & 16) ? 8 : 0);
#pragma unroll
  for (int t = 0; t < 4; ++t) {
    int j = k0 + t * 16 + ncl;
#pragma unroll
    for (int r = 0; r < 8; ++r) {
      int i = ibase + r;
      if (i < S && j < S)
        base[(size_t)i * S + j] = acc[t][r] + (j > i ? NEG : 0.0f);
    }
  }
}

// ---------------------------------------------------------------------------
// Row softmax over qk (len 1500 fp32) -> f16 w rows with stride SP=1536,
// cols [1500,1536) zeroed so the wv K-loop is tail-free.  12*128 == SP.
// ---------------------------------------------------------------------------
__global__ __launch_bounds__(128)
void k_softmax(const float* __restrict__ qk, f16* __restrict__ wh) {
  using namespace cfg;
  const int rid = blockIdx.x;            // over B*H*S rows
  const int tid = threadIdx.x;
  const float* row = qk + (size_t)rid * S;
  __shared__ float red[128];

  float vals[12];
  float m = -3.4e38f;
#pragma unroll
  for (int i = 0; i < 12; ++i) {
    int j = tid + i * 128;
    float v = (j < S) ? row[j] : -3.4e38f;
    vals[i] = v;
    m = fmaxf(m, v);
  }
  red[tid] = m;
  __syncthreads();
  for (int s = 64; s > 0; s >>= 1) {
    if (tid < s) red[tid] = fmaxf(red[tid], red[tid + s]);
    __syncthreads();
  }
  m = red[0];
  __syncthreads();

  float sum = 0.f;
#pragma unroll
  for (int i = 0; i < 12; ++i) {
    int j = tid + i * 128;
    float e = (j < S) ? __expf(vals[i] - m) : 0.f;
    vals[i] = e;
    sum += e;
  }
  red[tid] = sum;
  __syncthreads();
  for (int s = 64; s > 0; s >>= 1) {
    if (tid < s) red[tid] += red[tid + s];
    __syncthreads();
  }
  float inv = 1.0f / red[0];

  f16* wrow = wh + (size_t)rid * SP;
#pragma unroll
  for (int i = 0; i < 12; ++i) {
    int j = tid + i * 128;
    wrow[j] = (j < S) ? (f16)(vals[i] * inv) : (f16)0.f;
  }
}

// ---------------------------------------------------------------------------
// wv[b,s,h,:] = sum_k w[b,h,s,k] * vT[b,h,:,k]   (f16 out, [B,S,H,hd] layout)
// grid (qtile, b*H); 4 waves/block; 64 rows x 64 dims; K step 64 over SP.
// ---------------------------------------------------------------------------
__global__ __launch_bounds__(128)
void k_attn_wv(const f16* __restrict__ wh, const f16* __restrict__ vT,
               f16* __restrict__ wvh) {
  using namespace cfg;
  __shared__ f16 Ps[64][80];   // probs tile [qrow][k]
  __shared__ f16 Vs[64][80];   // vT tile    [d][k]

  const int tid  = threadIdx.x;
  const int lane = tid & 31;
  const int wid  = tid >> 5;
  const int q0 = blockIdx.x * 64;
  const int bh = blockIdx.y;
  const int b  = bh / H, h = bh % H;
  const f16* wbase  = wh + ((size_t)bh * S + q0) * SP;
  const f16* vtbase = vT + (size_t)bh * HD * SP;

  v8f acc[4];
#pragma unroll
  for (int t = 0; t < 4; ++t) acc[t] = (v8f){0.f, 0.f, 0.f, 0.f, 0.f, 0.f, 0.f, 0.f};

  const int kb  = (lane & 16) ? 8 : 0;
  const int kbb = (lane & 16) ? 16 : 0;
  const int arow = wid * 16 + (lane & 15);
  const int ncl  = lane & 15;

  for (int k0 = 0; k0 < SP; k0 += 64) {
#pragma unroll
    for (int it = 0; it < 4; ++it) {
      int idx = tid + it * 128;
      int r = idx >> 3, c = idx & 7;
      *(uint4*)&Ps[r][8 * c] = *(const uint4*)(wbase + (size_t)r * SP + k0 + 8 * c);
      *(uint4*)&Vs[r][8 * c] = *(const uint4*)(vtbase + (size_t)r * SP + k0 + 8 * c);
    }
    __syncthreads();

#pragma unroll
    for (int step = 0; step < 2; ++step) {
      int ko = step * 32;
      v16h a = cat16(*(const v8h*)&Ps[arow][ko + kbb],
                     *(const v8h*)&Ps[arow][ko + kbb + 8]);
#pragma unroll
      for (int t = 0; t < 4; ++t) {
        v16h bf = *(const v16h*)&Vs[t * 16 + ncl][ko + kbb];
        acc[t] = WMMA_F16(a, bf, acc[t]);
      }
    }
    __syncthreads();
  }

  const int sbase = q0 + wid * 16 + ((lane & 16) ? 8 : 0);
#pragma unroll
  for (int t = 0; t < 4; ++t) {
    int n = t * 16 + ncl;
#pragma unroll
    for (int r = 0; r < 8; ++r) {
      int s = sbase + r;
      if (s < S) wvh[(((size_t)b * S + s) * H + h) * HD + n] = (f16)acc[t][r];
    }
  }
}

// ---------------------------------------------------------------------------
// Host launcher
// ---------------------------------------------------------------------------
extern "C" void kernel_launch(void* const* d_in, const int* in_sizes, int n_in,
                              void* d_out, int out_size, void* d_ws, size_t ws_size,
                              hipStream_t stream) {
  using namespace cfg;
  (void)in_sizes; (void)n_in; (void)out_size; (void)ws_size;

  const float* x  = (const float*)d_in[0];
  // d_in[1] = mask: triu(NEG,k=1) — reproduced analytically in k_attn_qk.
  const float* Wq = (const float*)d_in[2];
  const float* bq = (const float*)d_in[3];
  const float* Wk = (const float*)d_in[4];
  const float* Wv = (const float*)d_in[5];
  const float* bv = (const float*)d_in[6];
  const float* Wo = (const float*)d_in[7];
  const float* bo = (const float*)d_in[8];

  float* out_f = (float*)d_out;                       // [B,S,D]
  float* qk_f  = out_f + (size_t)BS * D;              // [B,H,S,S]

  // workspace carve-up (256B aligned); row-padded so staging needs no guards
  char* ws = (char*)d_ws;
  size_t off = 0;
  auto carve = [&](size_t bytes) -> char* {
    char* p = ws + off;
    off = (off + bytes + 255) & ~(size_t)255;
    return p;
  };
  const size_t rowsBSp = (size_t)BS + PAD_ROWS;
  f16* x_h  = (f16*)carve(rowsBSp * D * 2);
  f16* Wq_h = (f16*)carve((size_t)D * D * 2);
  f16* Wk_h = (f16*)carve((size_t)D * D * 2);
  f16* Wv_h = (f16*)carve((size_t)D * D * 2);
  f16* Wo_h = (f16*)carve((size_t)D * D * 2);
  f16* q_h  = (f16*)carve(rowsBSp * D * 2);
  f16* k_h  = (f16*)carve(rowsBSp * D * 2);
  f16* v_h  = (f16*)carve(rowsBSp * D * 2);
  f16* vT_h = (f16*)carve((size_t)B * H * HD * SP * 2);
  f16* wv_h = (f16*)carve(rowsBSp * D * 2);
  f16* w_h  = (f16*)carve(((size_t)B * H * S + PAD_ROWS) * SP * 2);

  // 1) fp32 -> f16 conversions
  {
    int n = BS * D;
    k_f32_to_f16<<<(n + 255) / 256, 256, 0, stream>>>(x, x_h, n);
    n = D * D;
    dim3 g((n + 255) / 256);
    k_f32_to_f16<<<g, 256, 0, stream>>>(Wq, Wq_h, n);
    k_f32_to_f16<<<g, 256, 0, stream>>>(Wk, Wk_h, n);
    k_f32_to_f16<<<g, 256, 0, stream>>>(Wv, Wv_h, n);
    k_f32_to_f16<<<g, 256, 0, stream>>>(Wo, Wo_h, n);
  }

  // 2) projections: q = (xWq^T + bq)*scale, k = (xWk^T)*scale, v = xWv^T + bv
  {
    dim3 grid((BS + 63) / 64, D / 64);
    dim3 block(128);
    k_gemm_xWt<<<grid, block, 0, stream>>>(x_h, Wq_h, bq, SCALE, q_h, nullptr, BS, D, D);
    k_gemm_xWt<<<grid, block, 0, stream>>>(x_h, Wk_h, nullptr, SCALE, k_h, nullptr, BS, D, D);
    k_gemm_xWt<<<grid, block, 0, stream>>>(x_h, Wv_h, bv, 1.0f, v_h, nullptr, BS, D, D);
  }

  // 2b) transpose v for the wv GEMM (zero-padded keys)
  {
    size_t n = (size_t)B * H * HD * SP;
    k_transpose_v<<<(unsigned)((n + 255) / 256), 256, 0, stream>>>(v_h, vT_h);
  }

  // 3) qk = q.k^T + causal mask  (fp32, second output region)
  {
    dim3 grid((S + 63) / 64, (S + 63) / 64, B * H);
    k_attn_qk<<<grid, dim3(128), 0, stream>>>(q_h, k_h, qk_f);
  }

  // 4) softmax rows -> f16 w (padded stride)
  k_softmax<<<B * H * S, 128, 0, stream>>>(qk_f, w_h);

  // 5) wv = w @ v  -> f16 [B,S,H,hd]
  {
    dim3 grid((S + 63) / 64, B * H);
    k_attn_wv<<<grid, dim3(128), 0, stream>>>(w_h, vT_h, wv_h);
  }

  // 6) out = wv @ Wo^T + bo  (fp32, first output region)
  {
    dim3 grid((BS + 63) / 64, D / 64);
    k_gemm_xWt<<<grid, dim3(128), 0, stream>>>(wv_h, Wo_h, bo, 1.0f, nullptr, out_f, BS, D, D);
  }
}
